// ProbSparseAttention_69801808495148
// MI455X (gfx1250) — compile-verified, matrix operand order checked by reading, not run on hardware
//
#include <hip/hip_runtime.h>
#include <hip/hip_bf16.h>

// ---------------------------------------------------------------------------
// Types for CDNA5 WMMA (wave32): bf16 A/B fragments, f32 accumulator
// ---------------------------------------------------------------------------
typedef __attribute__((ext_vector_type(16))) __bf16 v16bf;
typedef __attribute__((ext_vector_type(8)))  __bf16 v8bf;
typedef __attribute__((ext_vector_type(8)))  float  v8f;
typedef __attribute__((ext_vector_type(4)))  float  v4f;
typedef int v4i_gcc __attribute__((vector_size(16)));   // matches builtin param type

#define D_MODEL 1024
#define N_HEADS 16
#define HEAD_DIM 64
#define SEQ_L 4096
#define BATCH 4
#define U_TOP 45
#define BH (BATCH * N_HEADS)
#define M_ROWS (BATCH * SEQ_L)

#define TILE_M 128
#define TILE_N 128
#define TILE_K 32
#define A_STRIDE 36   // fp32 elems per A LDS row (144B, 16B aligned)
#define B_STRIDE 40   // bf16 elems per B LDS row (80B, 16B aligned)

// Async global->LDS path (ASYNCcnt), guarded so the file compiles either way.
#if defined(__HIP_DEVICE_COMPILE__) && \
    __has_builtin(__builtin_amdgcn_global_load_async_to_lds_b128) && \
    __has_builtin(__builtin_amdgcn_s_wait_asynccnt)
#define USE_ASYNC_LDS 1
#else
#define USE_ASYNC_LDS 0
#endif

// ---------------------------------------------------------------------------
// Kernel 1: split-bf16 WMMA GEMM   Y = X(MxK) * W(KxN) + bias
//   split=1: 3-term bf16x3 (hi*hi + hi*lo + lo*hi)  ~ fp32 accuracy
//   split=0: 1-term bf16
//   mode 0: Y row-major (m,n);  mode 1: head-major Y[((b*16+h)*4096+l)*64+d]
// 256 threads = 8 waves; WG tile 128x128; wave tile 64x32 (4x2 WMMA tiles).
// ---------------------------------------------------------------------------
__global__ __launch_bounds__(256)
void gemm_bf16x3_wmma(const float* __restrict__ X, const float* __restrict__ W,
                      const float* __restrict__ bias, float* __restrict__ Y,
                      int M, int N, int K, int mode, int split)
{
    __shared__ __align__(16) float  Af [TILE_M * A_STRIDE];  // fp32 staging (18KB)
    __shared__ __align__(16) __bf16 Bhi[TILE_N * B_STRIDE];  // 10KB, transposed [n][k]
    __shared__ __align__(16) __bf16 Blo[TILE_N * B_STRIDE];  // 10KB

    const int tid  = threadIdx.x;
    const int lane = tid & 31;
    const int wave = tid >> 5;            // 0..7
    const int wm   = (wave >> 2) * 64;    // 2 wave-rows of 64
    const int wn   = (wave & 3) * 32;     // 4 wave-cols of 32
    const int m0   = blockIdx.x * TILE_M;
    const int n0   = blockIdx.y * TILE_N;
    const int halfsel = lane >> 4;        // 0: K{0..7,16..23}  1: K{8..15,24..31}
    const int l15     = lane & 15;

    v8f c[4][2];
    const v8f vzero = {0.f, 0.f, 0.f, 0.f, 0.f, 0.f, 0.f, 0.f};
#pragma unroll
    for (int i = 0; i < 4; ++i)
#pragma unroll
        for (int j = 0; j < 2; ++j) c[i][j] = vzero;

    for (int k0 = 0; k0 < K; k0 += TILE_K) {
        // ---- prefetch next K tile (global_prefetch_b8) ----
        if (k0 + TILE_K < K) {
            __builtin_prefetch(X + (size_t)(m0 + (tid >> 1)) * K + (k0 + TILE_K) + (tid & 1) * 16, 0, 1);
            __builtin_prefetch(W + (size_t)(k0 + TILE_K + (tid >> 4)) * N + n0 + (tid & 15) * 8, 0, 1);
        }

        // ---- A tile (128x32 fp32): async DMA to LDS when available ----
#pragma unroll
        for (int i = 0; i < 4; ++i) {
            const int cix = tid + i * 256;      // 1024 x 16B chunks
            const int r = cix >> 3, q = cix & 7;
            const float* g = X + (size_t)(m0 + r) * K + k0 + q * 4;
            float*       l = Af + r * A_STRIDE + q * 4;
#if USE_ASYNC_LDS
            __builtin_amdgcn_global_load_async_to_lds_b128(
                (__attribute__((address_space(1))) v4i_gcc*)g,
                (__attribute__((address_space(3))) v4i_gcc*)l, 0, 0);
#else
            *(v4f*)l = *(const v4f*)g;
#endif
        }

        // ---- B tile (32x128 fp32) -> transposed hi/lo bf16 LDS ----
#pragma unroll
        for (int i = 0; i < 16; ++i) {
            const int cix = tid + i * 256;      // 4096 elems, n fastest (coalesced)
            const int kk = cix >> 7, nn = cix & 127;
            const float w = W[(size_t)(k0 + kk) * N + n0 + nn];
            const __bf16 h = (__bf16)w;
            Bhi[nn * B_STRIDE + kk] = h;
            Blo[nn * B_STRIDE + kk] = (__bf16)(w - (float)h);
        }

#if USE_ASYNC_LDS
        __builtin_amdgcn_s_wait_asynccnt(0);
#endif
        __syncthreads();

        // ---- build A fragments: read fp32, split hi/lo (VALU co-execs w/ WMMA) ----
        v16bf ahi[4], alo[4];
#pragma unroll
        for (int i = 0; i < 4; ++i) {
            const float* p = Af + (wm + i * 16 + l15) * A_STRIDE + halfsel * 8;
            float t[16];
            *(v4f*)&t[0]  = *(const v4f*)(p +  0);
            *(v4f*)&t[4]  = *(const v4f*)(p +  4);
            *(v4f*)&t[8]  = *(const v4f*)(p + 16);
            *(v4f*)&t[12] = *(const v4f*)(p + 20);
#pragma unroll
            for (int e = 0; e < 16; ++e) {
                const __bf16 h = (__bf16)t[e];
                ahi[i][e] = h;
                alo[i][e] = (__bf16)(t[e] - (float)h);
            }
        }
        // ---- B fragments (already bf16 hi/lo in LDS) ----
        v16bf bhi[2], blo[2];
#pragma unroll
        for (int j = 0; j < 2; ++j) {
            const __bf16* p = Bhi + (wn + j * 16 + l15) * B_STRIDE + halfsel * 8;
            v8bf lo = *(const v8bf*)p;
            v8bf hi = *(const v8bf*)(p + 16);
            bhi[j] = __builtin_shufflevector(lo, hi, 0,1,2,3,4,5,6,7,8,9,10,11,12,13,14,15);
            const __bf16* q = Blo + (wn + j * 16 + l15) * B_STRIDE + halfsel * 8;
            v8bf lo2 = *(const v8bf*)q;
            v8bf hi2 = *(const v8bf*)(q + 16);
            blo[j] = __builtin_shufflevector(lo2, hi2, 0,1,2,3,4,5,6,7,8,9,10,11,12,13,14,15);
        }

        // ---- WMMA: 8 (1-term) or 24 (3-term) per K step per wave ----
#pragma unroll
        for (int j = 0; j < 2; ++j) {
#pragma unroll
            for (int i = 0; i < 4; ++i) {
                c[i][j] = __builtin_amdgcn_wmma_f32_16x16x32_bf16(
                    false, ahi[i], false, bhi[j], (short)0, c[i][j], false, false);
                if (split) {
                    c[i][j] = __builtin_amdgcn_wmma_f32_16x16x32_bf16(
                        false, ahi[i], false, blo[j], (short)0, c[i][j], false, false);
                    c[i][j] = __builtin_amdgcn_wmma_f32_16x16x32_bf16(
                        false, alo[i], false, bhi[j], (short)0, c[i][j], false, false);
                }
            }
        }
        __syncthreads();
    }

    // ---- store: lane holds (r -> M = base + halfsel*8 + r, N = base + l15) ----
#pragma unroll
    for (int i = 0; i < 4; ++i) {
#pragma unroll
        for (int j = 0; j < 2; ++j) {
            const int n  = n0 + wn + j * 16 + l15;
            const float bv = bias[n];
#pragma unroll
            for (int r = 0; r < 8; ++r) {
                const int m = m0 + wm + i * 16 + halfsel * 8 + r;
                const float val = c[i][j][r] + bv;
                if (mode == 0) {
                    Y[(size_t)m * N + n] = val;
                } else {
                    const int b_ = m >> 12, l_ = m & 4095;
                    const int h_ = n >> 6,  d_ = n & 63;
                    Y[(((size_t)(b_ * N_HEADS + h_) * SEQ_L) + l_) * HEAD_DIM + d_] = val;
                }
            }
        }
    }
}

// ---------------------------------------------------------------------------
// Kernel 2: sampled sparsity metric  M(bh,l) = scale*(max_s q.k_s - mean_s q.k_s)
// ---------------------------------------------------------------------------
__global__ __launch_bounds__(256)
void sample_scores(const float* __restrict__ Q, const float* __restrict__ Kh,
                   const int* __restrict__ sidx, float* __restrict__ Mout)
{
    __shared__ float Ks[U_TOP * HEAD_DIM];
    const int bh = blockIdx.y;
    const int l  = blockIdx.x * 256 + threadIdx.x;

    for (int i = threadIdx.x; i < U_TOP * HEAD_DIM; i += 256) {
        const int s = i >> 6, d = i & 63;
        Ks[i] = Kh[((size_t)bh * SEQ_L + sidx[s]) * HEAD_DIM + d];
    }
    __syncthreads();

    float q[HEAD_DIM];
    const float* qp = Q + ((size_t)bh * SEQ_L + l) * HEAD_DIM;
#pragma unroll
    for (int d = 0; d < HEAD_DIM; ++d) q[d] = qp[d];

    float mx = -1e30f, sum = 0.f;
    for (int s = 0; s < U_TOP; ++s) {
        const float* kp = Ks + s * HEAD_DIM;
        float dot = 0.f;
#pragma unroll
        for (int d = 0; d < HEAD_DIM; ++d) dot += q[d] * kp[d];
        mx = fmaxf(mx, dot);
        sum += dot;
    }
    const float scale = 0.125f; // 1/sqrt(64)
    Mout[(size_t)bh * SEQ_L + l] = scale * (mx - sum * (1.f / (float)U_TOP));
}

// ---------------------------------------------------------------------------
// Kernel 3: iterative top-45 argmax per (b,h) row of M
// ---------------------------------------------------------------------------
__global__ __launch_bounds__(256)
void topk45(const float* __restrict__ Mv, int* __restrict__ topIdx)
{
    __shared__ float vals[SEQ_L];
    __shared__ float bv[256];
    __shared__ int   bi[256];
    const int bh = blockIdx.x;

    for (int i = threadIdx.x; i < SEQ_L; i += 256)
        vals[i] = Mv[(size_t)bh * SEQ_L + i];
    __syncthreads();

    for (int t = 0; t < U_TOP; ++t) {
        float best = -1e30f; int bidx = 0;
        for (int i = threadIdx.x; i < SEQ_L; i += 256)
            if (vals[i] > best) { best = vals[i]; bidx = i; }
        bv[threadIdx.x] = best; bi[threadIdx.x] = bidx;
        __syncthreads();
        for (int s = 128; s > 0; s >>= 1) {
            if (threadIdx.x < s && bv[threadIdx.x + s] > bv[threadIdx.x]) {
                bv[threadIdx.x] = bv[threadIdx.x + s];
                bi[threadIdx.x] = bi[threadIdx.x + s];
            }
            __syncthreads();
        }
        if (threadIdx.x == 0) {
            topIdx[bh * U_TOP + t] = bi[0];
            vals[bi[0]] = -1e30f;
        }
        __syncthreads();
    }
}

// ---------------------------------------------------------------------------
// Kernel 4: full softmax attention for the 45 selected queries per head
// one block per (u, bh); V re-reads hit the 192MB L2 (V is 64MB total)
// ---------------------------------------------------------------------------
__global__ __launch_bounds__(256)
void sparse_attn(const float* __restrict__ Q, const float* __restrict__ Kh,
                 const float* __restrict__ V, const int* __restrict__ topIdx,
                 float* __restrict__ outRed)
{
    __shared__ float sc[SEQ_L];
    __shared__ float q[HEAD_DIM];
    __shared__ float red[256];
    __shared__ float outv[HEAD_DIM];
    const int u  = blockIdx.x;
    const int bh = blockIdx.y;
    const int lu = topIdx[bh * U_TOP + u];

    if (threadIdx.x < HEAD_DIM) {
        q[threadIdx.x]    = Q[((size_t)bh * SEQ_L + lu) * HEAD_DIM + threadIdx.x];
        outv[threadIdx.x] = 0.f;
    }
    __syncthreads();

    const float scale = 0.125f;
    float lmax = -1e30f;
    for (int l = threadIdx.x; l < SEQ_L; l += 256) {
        const float* kp = Kh + ((size_t)bh * SEQ_L + l) * HEAD_DIM;
        float dot = 0.f;
#pragma unroll
        for (int d = 0; d < HEAD_DIM; ++d) dot += q[d] * kp[d];
        dot *= scale;
        sc[l] = dot;
        lmax = fmaxf(lmax, dot);
    }
    red[threadIdx.x] = lmax; __syncthreads();
    for (int s = 128; s > 0; s >>= 1) {
        if (threadIdx.x < s) red[threadIdx.x] = fmaxf(red[threadIdx.x], red[threadIdx.x + s]);
        __syncthreads();
    }
    const float mx = red[0];
    __syncthreads();

    float lsum = 0.f;
    for (int l = threadIdx.x; l < SEQ_L; l += 256) {
        const float e = __expf(sc[l] - mx);
        sc[l] = e;
        lsum += e;
    }
    red[threadIdx.x] = lsum; __syncthreads();
    for (int s = 128; s > 0; s >>= 1) {
        if (threadIdx.x < s) red[threadIdx.x] += red[threadIdx.x + s];
        __syncthreads();
    }
    const float inv = 1.f / red[0];

    float acc[HEAD_DIM];
#pragma unroll
    for (int d = 0; d < HEAD_DIM; ++d) acc[d] = 0.f;
    for (int l = threadIdx.x; l < SEQ_L; l += 256) {
        const float w = sc[l];
        const float* vp = V + ((size_t)bh * SEQ_L + l) * HEAD_DIM;
#pragma unroll
        for (int d = 0; d < HEAD_DIM; ++d) acc[d] += w * vp[d];
    }
#pragma unroll
    for (int d = 0; d < HEAD_DIM; ++d) atomicAdd(&outv[d], acc[d]);
    __syncthreads();
    if (threadIdx.x < HEAD_DIM)
        outRed[((size_t)bh * U_TOP + u) * HEAD_DIM + threadIdx.x] = outv[threadIdx.x] * inv;
}

// ---------------------------------------------------------------------------
// Kernel 5: V mean over L per (bh, d)
// ---------------------------------------------------------------------------
__global__ __launch_bounds__(64)
void v_mean(const float* __restrict__ V, float* __restrict__ vm)
{
    const int bh = blockIdx.x;
    const int d  = threadIdx.x;
    float s = 0.f;
    for (int l = 0; l < SEQ_L; ++l)
        s += V[((size_t)bh * SEQ_L + l) * HEAD_DIM + d];
    vm[bh * HEAD_DIM + d] = s * (1.f / (float)SEQ_L);
}

// ---------------------------------------------------------------------------
// Kernel 6: fill assembled output (B,L,1024) with broadcast V-mean
// ---------------------------------------------------------------------------
__global__ __launch_bounds__(256)
void assemble_base(const float* __restrict__ vm, float* __restrict__ outb)
{
    const size_t idx = (size_t)blockIdx.x * 256 + threadIdx.x; // over B*L*1024
    const int    n   = (int)(idx & (D_MODEL - 1));             // h*64 + d
    const size_t m   = idx >> 10;                              // b*4096 + l
    const int    b   = (int)(m >> 12);
    const int    h   = n >> 6, d = n & 63;
    outb[idx] = vm[(b * N_HEADS + h) * HEAD_DIM + d];
}

// ---------------------------------------------------------------------------
// Kernel 7: scatter reduced-attention rows into assembled output
// ---------------------------------------------------------------------------
__global__ __launch_bounds__(64)
void scatter_red(const float* __restrict__ outRed, const int* __restrict__ topIdx,
                 float* __restrict__ outb)
{
    const int u  = blockIdx.x;
    const int bh = blockIdx.y;
    const int d  = threadIdx.x;
    const int b  = bh >> 4, h = bh & 15;
    const int l  = topIdx[bh * U_TOP + u];
    outb[((size_t)b * SEQ_L + l) * D_MODEL + h * HEAD_DIM + d] =
        outRed[((size_t)bh * U_TOP + u) * HEAD_DIM + d];
}

// ---------------------------------------------------------------------------
// Launch
// ---------------------------------------------------------------------------
extern "C" void kernel_launch(void* const* d_in, const int* in_sizes, int n_in,
                              void* d_out, int out_size, void* d_ws, size_t ws_size,
                              hipStream_t stream)
{
    const float* x    = (const float*)d_in[0];
    const float* Wq   = (const float*)d_in[1];
    const float* bq   = (const float*)d_in[2];
    const float* Wk   = (const float*)d_in[3];
    const float* bk   = (const float*)d_in[4];
    const float* Wv   = (const float*)d_in[5];
    const float* bv   = (const float*)d_in[6];
    const float* Wo   = (const float*)d_in[7];
    const float* bo   = (const float*)d_in[8];
    const int*   sidx = (const int*)d_in[9];
    float*       out  = (float*)d_out;

    char* ws = (char*)d_ws;
    const size_t SZ_MAT = (size_t)M_ROWS * D_MODEL * sizeof(float); // 64MB
    float* Qb   = (float*)(ws);                              // head-major Q
    float* Kb   = (float*)(ws + SZ_MAT);                     // head-major K
    float* Vb   = (float*)(ws + 2 * SZ_MAT);                 // head-major V
    float* Mb   = (float*)(ws + 3 * SZ_MAT);                 // sparsity metric (1MB)
    int*   tIdx = (int*)  (ws + 3 * SZ_MAT + (1u << 20));    // top indices
    float* oRed = (float*)(ws + 3 * SZ_MAT + (1u << 20) + (1u << 16));
    float* vm   = (float*)(ws + 3 * SZ_MAT + (1u << 20) + (1u << 16) + (1u << 20));
    float* outB = Qb; // Q is dead after sparse_attn; reuse its 64MB for assembly

    dim3 gGemm(M_ROWS / TILE_M, D_MODEL / TILE_N); // (128, 8)

    // Q/K/V projections: 3-term split-bf16 (feeds the discontinuous top-k)
    gemm_bf16x3_wmma<<<gGemm, 256, 0, stream>>>(x, Wq, bq, Qb, M_ROWS, D_MODEL, D_MODEL, 1, 1);
    gemm_bf16x3_wmma<<<gGemm, 256, 0, stream>>>(x, Wk, bk, Kb, M_ROWS, D_MODEL, D_MODEL, 1, 1);
    gemm_bf16x3_wmma<<<gGemm, 256, 0, stream>>>(x, Wv, bv, Vb, M_ROWS, D_MODEL, D_MODEL, 1, 1);

    sample_scores<<<dim3(SEQ_L / 256, BH), 256, 0, stream>>>(Qb, Kb, sidx, Mb);
    topk45<<<BH, 256, 0, stream>>>(Mb, tIdx);
    sparse_attn<<<dim3(U_TOP, BH), 256, 0, stream>>>(Qb, Kb, Vb, tIdx, oRed);
    v_mean<<<BH, 64, 0, stream>>>(Vb, vm);

    assemble_base<<<(M_ROWS * D_MODEL) / 256, 256, 0, stream>>>(vm, outB);
    scatter_red<<<dim3(U_TOP, BH), 64, 0, stream>>>(oRed, tIdx, outB);

    // Output projection: smooth map, 1-term bf16 suffices
    gemm_bf16x3_wmma<<<gGemm, 256, 0, stream>>>(outB, Wo, bo, out, M_ROWS, D_MODEL, D_MODEL, 0, 0);
}
